// ShortConv_43963285242541
// MI455X (gfx1250) — matrix operand order repaired and checked by reading, not compiled, not run on hardware
//
#include <hip/hip_runtime.h>
#include <hip/hip_bf16.h>
#include <stdint.h>

// ---------------------------------------------------------------------------
// ShortConv: in_proj GEMM (bf16 WMMA) -> gate+causal depthwise conv -> out GEMM
// B=4, T=4096, H=2048, L=4
// ---------------------------------------------------------------------------

typedef __attribute__((ext_vector_type(16))) __bf16 v16bf;
typedef __attribute__((ext_vector_type(8)))  __bf16 v8bf;
typedef __attribute__((ext_vector_type(8)))  float  v8f;
typedef __attribute__((ext_vector_type(4)))  float  v4f;

// ---- bf16 helpers via native hardware converts ----------------------------
__device__ __forceinline__ unsigned short f32_to_bf16(float f) {
    union { __bf16 b; unsigned short u; } v;
    v.b = (__bf16)f;                    // native cvt, RNE
    return v.u;
}

__device__ __forceinline__ float bf16_to_f32(unsigned short s) {
    union { unsigned short u; __bf16 b; } v;
    v.u = s;
    return (float)v.b;
}

// ---- conversion kernels ----------------------------------------------------
// 8 elements per thread: two b128 loads in, one b128 store out. n % 8 == 0.
__global__ __launch_bounds__(256)
void k_f32_to_bf16_v8(const float* __restrict__ in, unsigned short* __restrict__ out, long n8) {
    long i = (long)blockIdx.x * blockDim.x + threadIdx.x;
    if (i >= n8) return;
    const v4f* p = (const v4f*)(in + i * 8);
    v4f lo = p[0], hi = p[1];
    v8bf o;
#pragma unroll
    for (int e = 0; e < 4; ++e) { o[e] = (__bf16)lo[e]; o[e + 4] = (__bf16)hi[e]; }
    *(v8bf*)(out + i * 8) = o;
}

// in: (K x N) row-major f32 -> out: (N x K) row-major bf16.
// Each thread: 8 consecutive k for one n (strided reads, vector write). K%8==0.
__global__ __launch_bounds__(256)
void k_transpose_bf16(const float* __restrict__ in, unsigned short* __restrict__ out,
                      int K, int N) {
    long i = (long)blockIdx.x * blockDim.x + threadIdx.x;   // over N*(K/8)
    long total = (long)N * (K / 8);
    if (i >= total) return;
    int  k0 = (int)(i % (K / 8)) * 8;
    long nn = i / (K / 8);
    v8bf o;
#pragma unroll
    for (int j = 0; j < 8; ++j) o[j] = (__bf16)in[(long)(k0 + j) * N + nn];
    *(v8bf*)(out + nn * K + k0) = o;
}

// ---- WMMA fragment assembly from two 16B chunks ---------------------------
__device__ __forceinline__ v16bf frag2(const unsigned short* __restrict__ p0,
                                       const unsigned short* __restrict__ p1) {
    v8bf lo = *(const v8bf*)p0;
    v8bf hi = *(const v8bf*)p1;
    v16bf o;
#pragma unroll
    for (int i = 0; i < 8; ++i) { o[i] = lo[i]; o[i + 8] = hi[i]; }
    return o;
}

// ---- bf16 WMMA GEMM:  C(MxN) = A(MxK) * WT(NxK)^T -------------------------
// Block: 256 threads = 8 waves, wave grid 2(M) x 4(N).
// Wave tile 64x64 (4x4 fragments, 16 WMMAs / 8 fragment loads per K-step),
// block tile 128x256.  Fragments double-buffered: next K-step loads are issued
// before the current 16-WMMA burst so global latency overlaps compute in-wave.
//
// A fragment (16x32, row-major A, per ISA layout):
//   lanes 0-15 : row m+lane,    K=[k..k+7] and [k+16..k+23]
//   lanes 16-31: row m+lane-16, K=[k+8..k+15] and [k+24..k+31]
// B fragment (32x16 from N-major WT):
//   lanes 0-15 : col n+lane,    K=[k..k+7] and [k+16..k+23]
//   lanes 16-31: col n+lane-16, K=[k+8..k+15] and [k+24..k+31]
template <bool OUT_BF16>
__global__ __launch_bounds__(256)
void k_gemm_bf16(const unsigned short* __restrict__ A,
                 const unsigned short* __restrict__ Bt,
                 void* __restrict__ Cv,
                 int K, int ldc) {
    const int lane = threadIdx.x & 31;
    const int r    = lane & 15;
    const int hh   = lane >> 4;
    const int wave = threadIdx.x >> 5;
    const int wm   = wave >> 2;            // 0..1
    const int wn   = wave & 3;             // 0..3
    const int m0   = blockIdx.y * 128 + wm * 64;
    const int n0   = blockIdx.x * 256 + wn * 64;

    // per-lane fragment base pointers (chunk0); chunk1 at +16 (A) / +16 (B)
    const unsigned short* pa[4];
    const unsigned short* pb[4];
#pragma unroll
    for (int i = 0; i < 4; ++i) pa[i] = A  + (size_t)(m0 + 16 * i + r) * K + hh * 8;
#pragma unroll
    for (int j = 0; j < 4; ++j) pb[j] = Bt + (size_t)(n0 + 16 * j + r) * K + hh * 8;

    v8f acc[4][4];
#pragma unroll
    for (int i = 0; i < 4; ++i)
#pragma unroll
        for (int j = 0; j < 4; ++j)
#pragma unroll
            for (int e = 0; e < 8; ++e) acc[i][j][e] = 0.0f;

    v16bf a[4], b[4];
#pragma unroll
    for (int i = 0; i < 4; ++i) a[i] = frag2(pa[i], pa[i] + 16);
#pragma unroll
    for (int j = 0; j < 4; ++j) b[j] = frag2(pb[j], pb[j] + 16);

    for (int k = 32; k < K; k += 32) {
        v16bf an[4], bn[4];
#pragma unroll
        for (int i = 0; i < 4; ++i) an[i] = frag2(pa[i] + k, pa[i] + k + 16);
#pragma unroll
        for (int j = 0; j < 4; ++j) bn[j] = frag2(pb[j] + k, pb[j] + k + 16);
#pragma unroll
        for (int i = 0; i < 4; ++i)
#pragma unroll
            for (int j = 0; j < 4; ++j)
                acc[i][j] = __builtin_amdgcn_wmma_f32_16x16x32_bf16(
                    false, a[i], false, b[j], (short)0, acc[i][j], false, false);
#pragma unroll
        for (int i = 0; i < 4; ++i) a[i] = an[i];
#pragma unroll
        for (int j = 0; j < 4; ++j) b[j] = bn[j];
    }
#pragma unroll
    for (int i = 0; i < 4; ++i)
#pragma unroll
        for (int j = 0; j < 4; ++j)
            acc[i][j] = __builtin_amdgcn_wmma_f32_16x16x32_bf16(
                false, a[i], false, b[j], (short)0, acc[i][j], false, false);

    // C/D layout: VGPR e, lane l -> row e + 8*(l>>4), col l&15.
    // All index spaces fit in 32 bits (16384*6144 < 2^31): unsigned math only.
    const unsigned col    = (unsigned)r;
    const unsigned rowoff = (unsigned)hh * 8u;
#pragma unroll
    for (int i = 0; i < 4; ++i)
#pragma unroll
        for (int j = 0; j < 4; ++j)
#pragma unroll
            for (int e = 0; e < 8; ++e) {
                unsigned row = (unsigned)m0 + 16u * i + rowoff + (unsigned)e;
                unsigned cn  = (unsigned)n0 + 16u * j + col;
                unsigned idx = row * (unsigned)ldc + cn;
                float vv = acc[i][j][e];
                if (OUT_BF16)
                    ((unsigned short*)Cv)[idx] = f32_to_bf16(vv);
                else
                    ((float*)Cv)[idx] = vv;
            }
}

// ---- fused mask * (Bg*xg) gate + causal depthwise conv (L=4) + Cg mult ----
// bcx: (B*T) x 3H bf16 ; conv_w: H x 4 f32 ; mask: B*T bytes ; y: (B*T) x H bf16
__global__ __launch_bounds__(256)
void k_convgate(const unsigned short* __restrict__ bcx,
                const float* __restrict__ conv_w,
                const unsigned char* __restrict__ mask,
                unsigned short* __restrict__ y,
                int T, int H) {
    unsigned idx = blockIdx.x * blockDim.x + threadIdx.x;   // < 2^25, fits
    unsigned total = 4u * (unsigned)T * (unsigned)H;
    if (idx >= total) return;

    unsigned h   = idx % (unsigned)H;
    unsigned tok = idx / (unsigned)H;
    unsigned t   = tok % (unsigned)T;
    const unsigned H3 = 3u * (unsigned)H;

    float s = 0.0f;
#pragma unroll
    for (int k = 0; k < 4; ++k) {
        int tt = (int)t - 3 + k;
        if (tt >= 0) {
            unsigned tok2 = tok - 3u + (unsigned)k;         // same batch (tt >= 0)
            if (mask[tok2]) {
                float bg = bf16_to_f32(bcx[tok2 * H3 + h]);
                float xg = bf16_to_f32(bcx[tok2 * H3 + 2u * (unsigned)H + h]);
                s += conv_w[h * 4u + (unsigned)k] * (bg * xg);
            }
        }
    }
    float cg = bf16_to_f32(bcx[tok * H3 + (unsigned)H + h]);
    y[idx] = f32_to_bf16(cg * s);
}

// ---------------------------------------------------------------------------
extern "C" void kernel_launch(void* const* d_in, const int* in_sizes, int n_in,
                              void* d_out, int out_size, void* d_ws, size_t ws_size,
                              hipStream_t stream) {
    (void)in_sizes; (void)n_in; (void)out_size; (void)ws_size;

    const float*         x      = (const float*)d_in[0];
    const unsigned char* mask   = (const unsigned char*)d_in[1];
    const float*         W_in   = (const float*)d_in[2];
    const float*         conv_w = (const float*)d_in[3];
    const float*         W_out  = (const float*)d_in[4];
    float*               out    = (float*)d_out;

    const int  B = 4, T = 4096, H = 2048;
    const long MT = (long)B * T;          // 16384 rows
    const int  N1 = 3 * H;                // 6144
    const int  N2 = H;                    // 2048

    // workspace carve-out (bf16 buffers), 256B aligned
    uint8_t* ws  = (uint8_t*)d_ws;
    size_t   off = 0;
    auto carve = [&](size_t bytes) -> void* {
        off = (off + 255) & ~(size_t)255;
        void* p = ws + off;
        off += bytes;
        return p;
    };
    unsigned short* xbf   = (unsigned short*)carve(MT * H * 2);         //  64 MiB
    unsigned short* wtin  = (unsigned short*)carve((size_t)N1 * H * 2); //  24 MiB
    unsigned short* wtout = (unsigned short*)carve((size_t)N2 * H * 2); //   8 MiB
    unsigned short* bcx   = (unsigned short*)carve(MT * N1 * 2);        // 192 MiB
    unsigned short* ybf   = (unsigned short*)carve(MT * H * 2);         //  64 MiB

    // 1) x -> bf16 (vectorized, 8 elems/thread)
    {
        long n8 = (MT * H) / 8;
        k_f32_to_bf16_v8<<<dim3((unsigned)((n8 + 255) / 256)), 256, 0, stream>>>(x, xbf, n8);
    }
    // 2) W_in (H x 3H) -> WT_in (3H x H) bf16 ; W_out (H x H) -> WT_out bf16
    {
        long n = (long)N1 * (H / 8);
        k_transpose_bf16<<<dim3((unsigned)((n + 255) / 256)), 256, 0, stream>>>(W_in, wtin, H, N1);
        long m = (long)N2 * (H / 8);
        k_transpose_bf16<<<dim3((unsigned)((m + 255) / 256)), 256, 0, stream>>>(W_out, wtout, H, N2);
    }
    // 3) GEMM1: BCx = x @ W_in   (16384 x 6144, bf16 out)
    {
        dim3 grid(N1 / 256, (unsigned)(MT / 128));
        k_gemm_bf16<true><<<grid, 256, 0, stream>>>(xbf, wtin, (void*)bcx, H, N1);
    }
    // 4) gate + causal conv + Cg multiply -> y bf16
    {
        long n = MT * H;
        k_convgate<<<dim3((unsigned)((n + 255) / 256)), 256, 0, stream>>>(bcx, conv_w, mask, ybf, T, H);
    }
    // 5) GEMM2: out = y @ W_out  (16384 x 2048, f32 out)
    {
        dim3 grid(N2 / 256, (unsigned)(MT / 128));
        k_gemm_bf16<false><<<grid, 256, 0, stream>>>(ybf, wtout, (void*)out, H, N2);
    }
}
